// FusionBlock_38371237822901
// MI455X (gfx1250) — matrix operand order repaired and checked by reading, not compiled, not run on hardware
//
#include <hip/hip_runtime.h>
#include <hip/hip_bf16.h>
#include <math.h>

#define DIM   128
#define HEADS 4
#define EPSF  1e-5f

typedef __attribute__((ext_vector_type(16))) __bf16 v16bf;
typedef __attribute__((ext_vector_type(8)))  float  v8f;

__device__ __forceinline__ float gelu_exact(float x) {
    return 0.5f * x * (1.0f + erff(x * 0.70710678118654752440f));
}

// order-preserving float<->uint encoding for atomicMax on floats
__device__ __forceinline__ unsigned enc_f(float f) {
    unsigned u = __float_as_uint(f);
    return (u & 0x80000000u) ? ~u : (u | 0x80000000u);
}
__device__ __forceinline__ float dec_f(unsigned e) {
    return (e & 0x80000000u) ? __uint_as_float(e ^ 0x80000000u)
                             : __uint_as_float(~e);
}

// block-wide double sum; valid on thread 0 (blockDim.x == 256 -> 8 waves)
__device__ double block_reduce(double v, double* shm) {
    int lane = threadIdx.x & 31;
    int wid  = threadIdx.x >> 5;
#pragma unroll
    for (int o = 16; o > 0; o >>= 1) v += __shfl_xor(v, o, 32);
    if (lane == 0) shm[wid] = v;
    __syncthreads();
    double r = (threadIdx.x < (blockDim.x >> 5)) ? shm[threadIdx.x] : 0.0;
    if (wid == 0) {
#pragma unroll
        for (int o = 4; o > 0; o >>= 1) r += __shfl_xor(r, o, 32);
    }
    __syncthreads();
    return r;
}

// ---------------- init: zero accumulators & stats ----------------
__global__ void k_init(float* deg, float* hbuf, float* gbuf, float* denom,
                       unsigned* menc, double* stats, int N) {
    long i   = blockIdx.x * (long)blockDim.x + threadIdx.x;
    long tot = (long)N * DIM;
    if (i < tot) { hbuf[i] = 0.f; gbuf[i] = 0.f; }
    if (i < N) deg[i] = 0.f;
    if (i < (long)N * HEADS) { denom[i] = 0.f; menc[i] = 0u; }
    if (i < 8) stats[i] = 0.0;
}

// ---------------- transpose + bf16-convert both weight matrices ----------------
__global__ void k_prepw(const float* __restrict__ Wg, const float* __restrict__ Wl,
                        __bf16* __restrict__ wtg, __bf16* __restrict__ wtl) {
    int i = blockIdx.x * blockDim.x + threadIdx.x;   // 128*128 threads exactly
    int n = i >> 7, k = i & 127;                     // wt[n][k] = W[k][n]
    wtg[i] = (__bf16)Wg[k * DIM + n];
    wtl[i] = (__bf16)Wl[k * DIM + n];
}

// ---------------- GCN degree + rsqrt ----------------
__global__ void k_deg(const int* __restrict__ ei, const float* __restrict__ ew,
                      float* deg, int E) {
    int e = blockIdx.x * blockDim.x + threadIdx.x;
    if (e < E) atomicAdd(&deg[ei[E + e]], ew[e]);
}
__global__ void k_dis(const float* __restrict__ deg, float* dis, int N) {
    int i = blockIdx.x * blockDim.x + threadIdx.x;
    if (i < N) {
        float d = deg[i];
        dis[i] = d > 0.f ? rsqrtf(fmaxf(d, 1e-30f)) : 0.f;
    }
}

// ---------------- dual WMMA GEMM: xw = x@Wg, xl = x@Wl + b_l ----------------
// block = 256 threads = 8 waves; waves 0-3: GCN weights, waves 4-7: GAT lin.
// Each wave computes a 16x128 row-stripe with 8 accumulators (16x16 tiles).
__global__ __launch_bounds__(256)
void k_gemm_dual(const float* __restrict__ x, const __bf16* __restrict__ wtg,
                 const __bf16* __restrict__ wtl, const float* __restrict__ b_l,
                 float* __restrict__ xw, float* __restrict__ xl, int M) {
    const int lane = threadIdx.x & 31;
    const int wid  = threadIdx.x >> 5;
    const int mat  = wid >> 2;                 // 0 -> W_gcn, 1 -> W_l
    const int rt   = wid & 3;
    const int rowBase = blockIdx.x * 64 + rt * 16;
    const int r       = rowBase + (lane & 15);
    const bool rvalid = (r < M);
    const __bf16* wt  = mat ? wtl : wtg;
    float*        out = mat ? xl  : xw;

    v8f acc[8];
#pragma unroll
    for (int ct = 0; ct < 8; ++ct)
#pragma unroll
        for (int j = 0; j < 8; ++j) acc[ct][j] = 0.f;

#pragma unroll
    for (int kc = 0; kc < 4; ++kc) {
        // A fragment: 16x32 bf16. lane<16 holds K {0..15}+kc*32,
        // lane>=16 holds K {16..31}+kc*32 (contiguous, matching B).
        const int kbase = kc * 32 + ((lane >> 4) * 16);
        v16bf afrag;
        if (rvalid) {
            const float* xr = x + (size_t)r * DIM + kbase;
#pragma unroll
            for (int i = 0; i < 8; ++i) afrag[i]     = (__bf16)xr[i];
#pragma unroll
            for (int i = 0; i < 8; ++i) afrag[8 + i] = (__bf16)xr[8 + i];
        } else {
#pragma unroll
            for (int i = 0; i < 16; ++i) afrag[i] = (__bf16)0.f;
        }
        // B fragment: lane holds column n = ct*16+(lane&15); K contiguous in
        // transposed weights -> one 32-byte load.
        const int koff = kc * 32 + ((lane >> 4) * 16);
#pragma unroll
        for (int ct = 0; ct < 8; ++ct) {
            const int n = ct * 16 + (lane & 15);
            v16bf bfrag = *reinterpret_cast<const v16bf*>(wt + n * DIM + koff);
            acc[ct] = __builtin_amdgcn_wmma_f32_16x16x32_bf16(
                false, afrag, false, bfrag, (short)0, acc[ct], false, false);
        }
    }

    // store (C layout: vgpr j -> row j (lanes 0-15) / j+8 (lanes 16-31))
    const int rhalf = (lane >> 4) * 8;
#pragma unroll
    for (int ct = 0; ct < 8; ++ct) {
        const int col = ct * 16 + (lane & 15);
        const float bias = mat ? b_l[col] : 0.f;
#pragma unroll
        for (int j = 0; j < 8; ++j) {
            int row = rowBase + j + rhalf;
            if (row < M) out[(size_t)row * DIM + col] = acc[ct][j] + bias;
        }
    }
}

// ---------------- GCN message scatter: h[dst] += norm * xw[src] ----------------
__global__ void k_gcn_scatter(const int* __restrict__ ei, const float* __restrict__ ew,
                              const float* __restrict__ dis, const float* __restrict__ xw,
                              float* hbuf, int E) {
    int lane = threadIdx.x & 31;
    int e = blockIdx.x * (blockDim.x >> 5) + (threadIdx.x >> 5);
    if (e >= E) return;
    int s = ei[e], d = ei[E + e];
    float norm = dis[s] * ew[e] * dis[d];
    const float4 v = *reinterpret_cast<const float4*>(xw + (size_t)s * DIM + lane * 4);
    float* hb = hbuf + (size_t)d * DIM + lane * 4;
    atomicAdd(hb + 0, norm * v.x);
    atomicAdd(hb + 1, norm * v.y);
    atomicAdd(hb + 2, norm * v.z);
    atomicAdd(hb + 3, norm * v.w);
}

// ---------------- GATv2 logits + running segment-max ----------------
// one wave per edge; lane l owns channel l of each head (C == 32 == wave size)
__global__ void k_gat_logits(const int* __restrict__ ei, const float* __restrict__ ea,
                             const float* __restrict__ W_edge, const float* __restrict__ att,
                             const float* __restrict__ xl,
                             float* lbuf, unsigned* menc, int E) {
    int lane = threadIdx.x & 31;
    int e = blockIdx.x * (blockDim.x >> 5) + (threadIdx.x >> 5);
    if (e >= E) return;
    int s = ei[e], d = ei[E + e];
    float eav = ea[e];
    const float* xs = xl + (size_t)s * DIM;
    const float* xd = xl + (size_t)d * DIM;
#pragma unroll
    for (int h = 0; h < HEADS; ++h) {
        int ch = h * 32 + lane;
        float z = xd[ch] + xs[ch] + eav * W_edge[ch];
        z = z > 0.f ? z : 0.2f * z;              // leaky_relu(0.2)
        float p = z * att[ch];                   // att[h][lane]
#pragma unroll
        for (int o = 16; o > 0; o >>= 1) p += __shfl_xor(p, o, 32);
        if (lane == h) {
            lbuf[(size_t)e * HEADS + h] = p;
            atomicMax(&menc[(size_t)d * HEADS + h], enc_f(p));
        }
    }
}

// ---------------- exp(logit - max) + denom ----------------
__global__ void k_gat_expsum(const int* __restrict__ ei, const unsigned* __restrict__ menc,
                             float* lbuf, float* denom, int E) {
    long i = blockIdx.x * (long)blockDim.x + threadIdx.x;
    if (i >= (long)E * HEADS) return;
    int e = (int)(i >> 2), h = (int)(i & 3);
    int d = ei[E + e];
    float m = dec_f(menc[(size_t)d * HEADS + h]);
    float a = expf(lbuf[i] - m);
    lbuf[i] = a;
    atomicAdd(&denom[(size_t)d * HEADS + h], a);
}

// ---------------- GAT message scatter: g[dst] += a/denom * xl[src] ----------------
__global__ void k_gat_scatter(const int* __restrict__ ei, const float* __restrict__ lbuf,
                              const float* __restrict__ denom, const float* __restrict__ xl,
                              float* gbuf, int E) {
    int lane = threadIdx.x & 31;
    int e = blockIdx.x * (blockDim.x >> 5) + (threadIdx.x >> 5);
    if (e >= E) return;
    int s = ei[e], d = ei[E + e];
    int h = lane >> 3;                           // 4 channels/lane -> head = lane/8
    float coef = lbuf[(size_t)e * HEADS + h] /
                 (denom[(size_t)d * HEADS + h] + 1e-16f);
    const float4 v = *reinterpret_cast<const float4*>(xl + (size_t)s * DIM + lane * 4);
    float* gb = gbuf + (size_t)d * DIM + lane * 4;
    atomicAdd(gb + 0, coef * v.x);
    atomicAdd(gb + 1, coef * v.y);
    atomicAdd(gb + 2, coef * v.z);
    atomicAdd(gb + 3, coef * v.w);
}

// ---------------- add biases + global sum/sumsq for both LNs ----------------
__global__ void k_bias_stats(float* hbuf, float* gbuf, const float* __restrict__ b_gcn,
                             const float* __restrict__ b_gat, double* stats, long tot) {
    __shared__ double shm[8];
    long i = blockIdx.x * (long)blockDim.x + threadIdx.x;
    double sh = 0, sh2 = 0, sg = 0, sg2 = 0;
    if (i < tot) {
        int ch = (int)(i & (DIM - 1));
        float hv = hbuf[i] + b_gcn[ch]; hbuf[i] = hv;
        float gv = gbuf[i] + b_gat[ch]; gbuf[i] = gv;
        sh = hv; sh2 = (double)hv * hv;
        sg = gv; sg2 = (double)gv * gv;
    }
    double t;
    t = block_reduce(sh,  shm); if (threadIdx.x == 0) atomicAdd(&stats[0], t);
    t = block_reduce(sh2, shm); if (threadIdx.x == 0) atomicAdd(&stats[1], t);
    t = block_reduce(sg,  shm); if (threadIdx.x == 0) atomicAdd(&stats[2], t);
    t = block_reduce(sg2, shm); if (threadIdx.x == 0) atomicAdd(&stats[3], t);
}

// ---------------- normalize both branches, fuse, gather fused stats ----------------
__global__ void k_fuse(const float* __restrict__ x, const float* __restrict__ hbuf,
                       const float* __restrict__ gbuf,
                       const float* __restrict__ gcn_nw, const float* __restrict__ gcn_nb,
                       const float* __restrict__ gat_nw, const float* __restrict__ gat_nb,
                       const float* __restrict__ alpha, double* stats,
                       float* fbuf, long tot) {
    __shared__ double shm[8];
    long i = blockIdx.x * (long)blockDim.x + threadIdx.x;
    double inv = 1.0 / (double)tot;
    float mu_h = (float)(stats[0] * inv);
    float sd_h = sqrtf(fmaxf((float)(stats[1] * inv) - mu_h * mu_h, 0.f));
    float mu_g = (float)(stats[2] * inv);
    float sd_g = sqrtf(fmaxf((float)(stats[3] * inv) - mu_g * mu_g, 0.f));
    float a0 = alpha[0], a1 = alpha[1];
    float mx = fmaxf(a0, a1);
    float e0 = expf(a0 - mx), e1 = expf(a1 - mx);
    float w0 = e0 / (e0 + e1), w1 = e1 / (e0 + e1);
    double s = 0, s2 = 0;
    if (i < tot) {
        int ch = (int)(i & (DIM - 1));
        float hn = (hbuf[i] - mu_h) / (sd_h + EPSF) * gcn_nw[ch] + gcn_nb[ch];
        float gn = (gbuf[i] - mu_g) / (sd_g + EPSF) * gat_nw[ch] + gat_nb[ch];
        float xv = x[i];
        float f = w0 * (xv + gelu_exact(hn)) + w1 * (xv + gelu_exact(gn));
        fbuf[i] = f;
        s = f; s2 = (double)f * f;
    }
    double t;
    t = block_reduce(s,  shm); if (threadIdx.x == 0) atomicAdd(&stats[4], t);
    t = block_reduce(s2, shm); if (threadIdx.x == 0) atomicAdd(&stats[5], t);
}

// ---------------- final LN + GELU ----------------
__global__ void k_final(const float* __restrict__ fbuf, const float* __restrict__ out_nw,
                        const float* __restrict__ out_nb, const double* __restrict__ stats,
                        float* __restrict__ out, long tot) {
    long i = blockIdx.x * (long)blockDim.x + threadIdx.x;
    if (i >= tot) return;
    double inv = 1.0 / (double)tot;
    float mu = (float)(stats[4] * inv);
    float sd = sqrtf(fmaxf((float)(stats[5] * inv) - mu * mu, 0.f));
    int ch = (int)(i & (DIM - 1));
    float v = (fbuf[i] - mu) / (sd + EPSF) * out_nw[ch] + out_nb[ch];
    out[i] = gelu_exact(v);
}

extern "C" void kernel_launch(void* const* d_in, const int* in_sizes, int n_in,
                              void* d_out, int out_size, void* d_ws, size_t ws_size,
                              hipStream_t stream) {
    const float* x      = (const float*)d_in[0];
    const int*   ei     = (const int*)  d_in[1];
    const float* ew     = (const float*)d_in[2];
    const float* ea     = (const float*)d_in[3];
    const float* Wg     = (const float*)d_in[4];
    const float* b_gcn  = (const float*)d_in[5];
    const float* Wl     = (const float*)d_in[6];
    const float* b_l    = (const float*)d_in[7];
    const float* W_edge = (const float*)d_in[8];
    const float* att    = (const float*)d_in[9];
    const float* b_gat  = (const float*)d_in[10];
    const float* gcn_nw = (const float*)d_in[11];
    const float* gcn_nb = (const float*)d_in[12];
    const float* gat_nw = (const float*)d_in[13];
    const float* gat_nb = (const float*)d_in[14];
    const float* out_nw = (const float*)d_in[15];
    const float* out_nb = (const float*)d_in[16];
    const float* alpha  = (const float*)d_in[17];

    const int  N   = in_sizes[0] / DIM;
    const int  E   = in_sizes[2];
    const long tot = (long)N * DIM;

    char*  base = (char*)d_ws;
    size_t off  = 0;
    auto alloc = [&](size_t bytes) -> void* {
        void* p = base + off;
        off = (off + bytes + 255) & ~(size_t)255;
        return p;
    };
    __bf16*   wtg   = (__bf16*)  alloc((size_t)DIM * DIM * sizeof(__bf16));
    __bf16*   wtl   = (__bf16*)  alloc((size_t)DIM * DIM * sizeof(__bf16));
    float*    xw    = (float*)   alloc((size_t)tot * 4);
    float*    xl    = (float*)   alloc((size_t)tot * 4);
    float*    hbuf  = (float*)   alloc((size_t)tot * 4);
    float*    gbuf  = (float*)   alloc((size_t)tot * 4);
    float*    deg   = (float*)   alloc((size_t)N * 4);
    float*    dis   = (float*)   alloc((size_t)N * 4);
    unsigned* menc  = (unsigned*)alloc((size_t)N * HEADS * 4);
    float*    denom = (float*)   alloc((size_t)N * HEADS * 4);
    float*    lbuf  = (float*)   alloc((size_t)E * HEADS * 4);
    double*   stats = (double*)  alloc(8 * sizeof(double));
    float*    fbuf  = xw;   // xw dead after k_gcn_scatter; reuse as fused buffer

    const int TB   = 256;
    const int gTot = (int)((tot + TB - 1) / TB);
    const int gE   = (E + TB - 1) / TB;
    const int gEW  = (E + 7) / 8;              // 8 edge-waves per 256-thread block

    k_init<<<gTot, TB, 0, stream>>>(deg, hbuf, gbuf, denom, menc, stats, N);
    k_prepw<<<(DIM * DIM) / TB, TB, 0, stream>>>(Wg, Wl, wtg, wtl);
    k_deg<<<gE, TB, 0, stream>>>(ei, ew, deg, E);
    k_dis<<<(N + TB - 1) / TB, TB, 0, stream>>>(deg, dis, N);
    k_gemm_dual<<<(N + 63) / 64, 256, 0, stream>>>(x, wtg, wtl, b_l, xw, xl, N);
    k_gcn_scatter<<<gEW, 256, 0, stream>>>(ei, ew, dis, xw, hbuf, E);
    k_gat_logits<<<gEW, 256, 0, stream>>>(ei, ea, W_edge, att, xl, lbuf, menc, E);
    k_gat_expsum<<<(int)(((long)E * HEADS + TB - 1) / TB), TB, 0, stream>>>(ei, menc, lbuf, denom, E);
    k_gat_scatter<<<gEW, 256, 0, stream>>>(ei, lbuf, denom, xl, gbuf, E);
    k_bias_stats<<<gTot, TB, 0, stream>>>(hbuf, gbuf, b_gcn, b_gat, stats, tot);
    k_fuse<<<gTot, TB, 0, stream>>>(x, hbuf, gbuf, gcn_nw, gcn_nb, gat_nw, gat_nb,
                                    alpha, stats, fbuf, tot);
    k_final<<<gTot, TB, 0, stream>>>(fbuf, out_nw, out_nb, stats, (float*)d_out, tot);
}
